// VisualSelectedEmbeddingLayer_21036749815976
// MI455X (gfx1250) — matrix operand order repaired and checked by reading, not compiled, not run on hardware
//
#include <hip/hip_runtime.h>
#include <hip/hip_bf16.h>
#include <math.h>

// ---------------- types ----------------
typedef __attribute__((ext_vector_type(16))) __bf16       bf16x16;
typedef __attribute__((ext_vector_type(8)))  float        f32x8;
typedef __attribute__((ext_vector_type(4)))  float        f32x4;
typedef __attribute__((ext_vector_type(2)))  float        f32x2;
typedef __attribute__((ext_vector_type(4)))  unsigned int u32x4;
typedef __attribute__((ext_vector_type(2)))  unsigned int u32x2;

#define BATCH 64
#define LSEQ  577
#define DDIM  512
#define EDIM  1024
#define HDIM  512
#define KSEL  288                 // (577-1)*0.5
#define ROWS  (BATCH * KSEL)      // 18432

// ---------------- helpers ----------------
__device__ __forceinline__ unsigned short f2bf(float f) {
  union { float f; unsigned u; } x; x.f = f;
  unsigned r = x.u + 0x7FFFu + ((x.u >> 16) & 1u);  // round-to-nearest-even
  return (unsigned short)(r >> 16);
}

// A-matrix 16x32 bf16 fragment (ISA 7.12.2): lanes<16 carry K {k0..k0+7, k0+16..k0+23},
// lanes>=16 carry K {k0+8..k0+15, k0+24..k0+31}. row base passed in elements.
__device__ __forceinline__ bf16x16 load_fragA(const unsigned short* p, size_t rowbase,
                                              int k0, int hf) {
  union { bf16x16 v; u32x4 q[2]; } f;
  const size_t e0 = rowbase + (size_t)(k0 + hf * 8);
  f.q[0] = *(const u32x4*)(p + e0);
  f.q[1] = *(const u32x4*)(p + e0 + 16);
  return f.v;
}

// B-matrix 32x16 bf16 fragment: lane = column N,
// lanes<16 carry K {k0..k0+7, k0+16..k0+23}, lanes>=16 carry K {k0+8..k0+15, k0+24..k0+31}.
__device__ __forceinline__ bf16x16 load_fragB(const unsigned short* p, size_t rowbase,
                                              int k0, int hf) {
  union { bf16x16 v; u32x4 q[2]; } f;
  const size_t e0 = rowbase + (size_t)(k0 + hf * 8);
  f.q[0] = *(const u32x4*)(p + e0);
  f.q[1] = *(const u32x4*)(p + e0 + 16);
  return f.v;
}

__device__ __forceinline__ f32x8 wmma_bf16(bf16x16 a, bf16x16 b, f32x8 c) {
  return __builtin_amdgcn_wmma_f32_16x16x32_bf16(false, a, false, b, (short)0, c,
                                                 false, false);
}

// ---------------- K1: top-k + gather + l2norm -> bf16 feats ----------------
__global__ void __launch_bounds__(256)
k_topk_gather(const float* __restrict__ base, const float* __restrict__ atten,
              unsigned short* __restrict__ featsb) {
  __shared__ float s[LSEQ];
  __shared__ int   tos[KSEL];
  const int b = blockIdx.x;
  const float* arow = atten + (size_t)b * LSEQ * LSEQ;   // CLS row: atten[b,0,:]
  for (int j = threadIdx.x; j < LSEQ; j += 256)
    s[j] = (j == 0) ? -1.0f : arow[j];
  __syncthreads();
  // rank = number of elements strictly beating s[j] (jax top_k tie-break: lower idx wins)
  for (int j = threadIdx.x; j < LSEQ; j += 256) {
    const float v = s[j];
    int rank = 0;
    for (int i = 0; i < LSEQ; ++i) {
      const float u = s[i];
      rank += (u > v) || ((u == v) && (i < j));
    }
    if (rank < KSEL) tos[rank] = j;
  }
  __syncthreads();
  const int wid = threadIdx.x >> 5, lane = threadIdx.x & 31;
  for (int slot = wid; slot < KSEL; slot += 8) {   // one wave per token
    const int tok = tos[slot];
    const float* src = base + ((size_t)b * LSEQ + tok) * DDIM + lane * 16;
    float x[16];
    float acc2 = 0.0f;
    #pragma unroll
    for (int q = 0; q < 4; ++q) {
      const f32x4 v = *(const f32x4*)(src + q * 4);
      #pragma unroll
      for (int i = 0; i < 4; ++i) { x[q * 4 + i] = v[i]; acc2 += v[i] * v[i]; }
    }
    #pragma unroll
    for (int m = 1; m < 32; m <<= 1) acc2 += __shfl_xor(acc2, m, 32);
    const float inv = 1.0f / (sqrtf(acc2) + 1e-8f);
    unsigned pk[8];
    #pragma unroll
    for (int i = 0; i < 8; ++i)
      pk[i] = (unsigned)f2bf(x[2 * i] * inv) |
              ((unsigned)f2bf(x[2 * i + 1] * inv) << 16);
    unsigned short* dst = featsb + ((size_t)(b * KSEL + slot)) * DDIM + lane * 16;
    u32x4 q0 = {pk[0], pk[1], pk[2], pk[3]};
    u32x4 q1 = {pk[4], pk[5], pk[6], pk[7]};
    *(u32x4*)(dst) = q0;
    *(u32x4*)(dst + 8) = q1;
  }
}

// ---------------- K2: f32 -> bf16 weight convert ----------------
__global__ void __launch_bounds__(256)
k_f2bf(const float* __restrict__ src, unsigned short* __restrict__ dst, int n) {
  const int i = blockIdx.x * 256 + threadIdx.x;
  if (i < n) dst[i] = f2bf(src[i]);
}

// ---------------- K3: GEMM1  h = feats @ w0^T + b0  (f32 out) ----------------
// grid (ROWS/64, HDIM/128), 8 waves/block, each wave -> 32x32 C block.
__global__ void __launch_bounds__(256)
k_gemm1(const unsigned short* __restrict__ featsb, const unsigned short* __restrict__ w0b,
        const float* __restrict__ bias, float* __restrict__ h) {
  const int lane = threadIdx.x & 31, wid = threadIdx.x >> 5;
  const int hf = lane >> 4, l15 = lane & 15;
  const int mt  = blockIdx.x * 64  + (wid & 1) * 32;
  const int ntb = blockIdx.y * 128 + (wid >> 1) * 32;
  const size_t rowA0 = (size_t)(mt + l15) * DDIM;
  const size_t rowA1 = rowA0 + (size_t)16 * DDIM;
  const size_t rowB0 = (size_t)(ntb + l15) * DDIM;
  const size_t rowB1 = rowB0 + (size_t)16 * DDIM;
  f32x8 a00 = {}, a01 = {}, a10 = {}, a11 = {};
  for (int k = 0; k < DDIM; k += 32) {
    const bf16x16 A0 = load_fragA(featsb, rowA0, k, hf);
    const bf16x16 A1 = load_fragA(featsb, rowA1, k, hf);
    const bf16x16 B0 = load_fragB(w0b, rowB0, k, hf);
    const bf16x16 B1 = load_fragB(w0b, rowB1, k, hf);
    a00 = wmma_bf16(A0, B0, a00);
    a01 = wmma_bf16(A0, B1, a01);
    a10 = wmma_bf16(A1, B0, a10);
    a11 = wmma_bf16(A1, B1, a11);
  }
  const int mb = mt + hf * 8;          // C layout: VGPR r -> M=r (lanes<16), M=8+r (lanes>=16)
  const int n0 = ntb + l15;
  const float bz0 = bias[n0], bz1 = bias[n0 + 16];
  #pragma unroll
  for (int r = 0; r < 8; ++r) {
    h[(size_t)(mb + r)      * HDIM + n0]      = a00[r] + bz0;
    h[(size_t)(mb + r)      * HDIM + n0 + 16] = a01[r] + bz1;
    h[(size_t)(mb + 16 + r) * HDIM + n0]      = a10[r] + bz0;
    h[(size_t)(mb + 16 + r) * HDIM + n0 + 16] = a11[r] + bz1;
  }
}

// ---------------- K4a: per-chunk column sums (deterministic, no atomics) ----------------
__global__ void __launch_bounds__(256)
k_bnpart(const float* __restrict__ h, float* __restrict__ psum, float* __restrict__ psq) {
  const int c0 = threadIdx.x * 2;           // thread owns 2 columns -> coalesced rows
  float s0 = 0, s1 = 0, q0 = 0, q1 = 0;
  const size_t r0 = (size_t)blockIdx.x * 256;
  for (int r = 0; r < 256; ++r) {
    const f32x2 v = *(const f32x2*)(h + (r0 + r) * HDIM + c0);
    s0 += v[0]; s1 += v[1]; q0 += v[0] * v[0]; q1 += v[1] * v[1];
  }
  psum[blockIdx.x * HDIM + c0] = s0; psum[blockIdx.x * HDIM + c0 + 1] = s1;
  psq [blockIdx.x * HDIM + c0] = q0; psq [blockIdx.x * HDIM + c0 + 1] = q1;
}

// ---------------- K4b: finalize BN -> scale/shift ----------------
__global__ void __launch_bounds__(256)
k_bnfin(const float* __restrict__ psum, const float* __restrict__ psq,
        const float* __restrict__ gamma, const float* __restrict__ beta,
        float* __restrict__ ss) {
  const int c0 = threadIdx.x * 2;
  float s0 = 0, s1 = 0, q0 = 0, q1 = 0;
  for (int p = 0; p < ROWS / 256; ++p) {
    s0 += psum[p * HDIM + c0]; s1 += psum[p * HDIM + c0 + 1];
    q0 += psq [p * HDIM + c0]; q1 += psq [p * HDIM + c0 + 1];
  }
  const float invN = 1.0f / (float)ROWS;
  const float m0 = s0 * invN, m1 = s1 * invN;
  const float v0 = fmaxf(q0 * invN - m0 * m0, 0.0f);
  const float v1 = fmaxf(q1 * invN - m1 * m1, 0.0f);
  const float sc0 = gamma[c0]     * rsqrtf(v0 + 1e-5f);
  const float sc1 = gamma[c0 + 1] * rsqrtf(v1 + 1e-5f);
  ss[c0] = sc0;               ss[c0 + 1] = sc1;
  ss[HDIM + c0] = beta[c0] - m0 * sc0;
  ss[HDIM + c0 + 1] = beta[c0 + 1] - m1 * sc1;
}

// ---------------- K5: BN + ReLU + bf16 convert ----------------
__global__ void __launch_bounds__(256)
k_bnrelu(const float* __restrict__ h, const float* __restrict__ ss,
         unsigned short* __restrict__ hb) {
  const size_t i4 = (size_t)blockIdx.x * 256 + threadIdx.x;
  const size_t e = i4 * 4;
  const int c = (int)(e & (HDIM - 1));
  f32x4 v = *(const f32x4*)(h + e);
  const f32x4 sc = *(const f32x4*)(ss + c);
  const f32x4 sh = *(const f32x4*)(ss + HDIM + c);
  #pragma unroll
  for (int i = 0; i < 4; ++i) v[i] = fmaxf(v[i] * sc[i] + sh[i], 0.0f);
  u32x2 p;
  p[0] = (unsigned)f2bf(v[0]) | ((unsigned)f2bf(v[1]) << 16);
  p[1] = (unsigned)f2bf(v[2]) | ((unsigned)f2bf(v[3]) << 16);
  *(u32x2*)(hb + e) = p;
}

// ---------------- K6: fused GEMM2a+2b + max-over-tokens epilogue ----------------
// grid (BATCH, EDIM/128); wave wid -> 16-column strip; loops over 18 M-tiles.
__global__ void __launch_bounds__(256)
k_gemm2(const unsigned short* __restrict__ featsb, const unsigned short* __restrict__ fcwb,
        const unsigned short* __restrict__ hb,     const unsigned short* __restrict__ w1b,
        const float* __restrict__ fc_b, const float* __restrict__ b1,
        float* __restrict__ out) {
  const int lane = threadIdx.x & 31, wid = threadIdx.x >> 5;
  const int hf = lane >> 4, l15 = lane & 15;
  const int b  = blockIdx.x;
  const int nt = blockIdx.y * 8 + wid;
  const int n  = nt * 16 + l15;
  const size_t rowB = (size_t)n * DDIM;      // fc_w and w1 both K=512 row-major
  float rm = -INFINITY;
  for (int mt = 0; mt < KSEL / 16; ++mt) {
    const size_t rowA = (size_t)(b * KSEL + mt * 16 + l15) * DDIM;
    f32x8 acc = {};
    for (int k = 0; k < DDIM; k += 32) {      // feats @ fc_w^T
      const bf16x16 A  = load_fragA(featsb, rowA, k, hf);
      const bf16x16 Bf = load_fragB(fcwb,  rowB, k, hf);
      acc = wmma_bf16(A, Bf, acc);
    }
    for (int k = 0; k < HDIM; k += 32) {      // + h_relu @ w1^T
      const bf16x16 A  = load_fragA(hb,  rowA, k, hf);
      const bf16x16 Bf = load_fragB(w1b, rowB, k, hf);
      acc = wmma_bf16(A, Bf, acc);
    }
    float t = acc[0];
    #pragma unroll
    for (int r = 1; r < 8; ++r) t = fmaxf(t, acc[r]);
    rm = fmaxf(rm, t);                        // max over this tile's 8 rows (per half)
  }
  rm = fmaxf(rm, __shfl_xor(rm, 16, 32));     // combine M halves (lane n vs n+16)
  if (lane < 16)                              // bias is constant along max dim -> add after
    out[(size_t)b * EDIM + nt * 16 + lane] = rm + fc_b[n] + b1[n];
}

// ---------------- launcher ----------------
extern "C" void kernel_launch(void* const* d_in, const int* in_sizes, int n_in,
                              void* d_out, int out_size, void* d_ws, size_t ws_size,
                              hipStream_t stream) {
  (void)in_sizes; (void)n_in; (void)out_size; (void)ws_size;
  const float* base  = (const float*)d_in[0];
  const float* atten = (const float*)d_in[1];
  const float* fc_w  = (const float*)d_in[2];
  const float* fc_b  = (const float*)d_in[3];
  const float* w0    = (const float*)d_in[4];
  const float* b0    = (const float*)d_in[5];
  const float* gam   = (const float*)d_in[6];
  const float* bet   = (const float*)d_in[7];
  const float* w1    = (const float*)d_in[8];
  const float* b1    = (const float*)d_in[9];
  float* out = (float*)d_out;

  char* p = (char*)d_ws;
  unsigned short* featsb = (unsigned short*)p; p += (size_t)ROWS * DDIM * 2;
  unsigned short* w0b    = (unsigned short*)p; p += (size_t)HDIM * DDIM * 2;
  unsigned short* fcwb   = (unsigned short*)p; p += (size_t)EDIM * DDIM * 2;
  unsigned short* w1b    = (unsigned short*)p; p += (size_t)EDIM * HDIM * 2;
  float*          h      = (float*)p;          p += (size_t)ROWS * HDIM * 4;
  unsigned short* hb     = (unsigned short*)p; p += (size_t)ROWS * HDIM * 2;
  float*          psum   = (float*)p;          p += (size_t)(ROWS / 256) * HDIM * 4;
  float*          psq    = (float*)p;          p += (size_t)(ROWS / 256) * HDIM * 4;
  float*          ss     = (float*)p;          p += (size_t)2 * HDIM * 4;

  k_topk_gather<<<BATCH, 256, 0, stream>>>(base, atten, featsb);
  k_f2bf<<<(HDIM * DDIM + 255) / 256, 256, 0, stream>>>(w0, w0b, HDIM * DDIM);
  k_f2bf<<<(EDIM * DDIM + 255) / 256, 256, 0, stream>>>(fc_w, fcwb, EDIM * DDIM);
  k_f2bf<<<(EDIM * HDIM + 255) / 256, 256, 0, stream>>>(w1, w1b, EDIM * HDIM);
  k_gemm1<<<dim3(ROWS / 64, HDIM / 128), 256, 0, stream>>>(featsb, w0b, b0, h);
  k_bnpart<<<ROWS / 256, 256, 0, stream>>>(h, psum, psq);
  k_bnfin<<<1, 256, 0, stream>>>(psum, psq, gam, bet, ss);
  k_bnrelu<<<(int)(((size_t)ROWS * HDIM / 4 + 255) / 256), 256, 0, stream>>>(h, ss, hb);
  k_gemm2<<<dim3(BATCH, EDIM / 128), 256, 0, stream>>>(featsb, fcwb, hb, w1b, fc_b, b1, out);
}
